// MultiComponentLocalField_v2_16587163697864
// MI455X (gfx1250) — matrix operand order repaired and checked by reading, not compiled, not run on hardware
//
#include <hip/hip_runtime.h>
#include <hip/hip_bf16.h>
#include <stdint.h>

typedef __attribute__((ext_vector_type(8)))  float        v8f;
typedef __attribute__((ext_vector_type(16))) __bf16       v16bf;
typedef __attribute__((ext_vector_type(4)))  unsigned int v4u;
typedef __attribute__((ext_vector_type(8)))  int          v8i;
typedef __attribute__((ext_vector_type(4)))  int          v4i;

union FragBF16 {
    unsigned int u[8];
    v16bf        v;
};

__device__ __forceinline__ unsigned short f2bf(float f) {
    unsigned int u = __float_as_uint(f);
    unsigned int r = u + 0x7FFFu + ((u >> 16) & 1u);   // round-to-nearest-even
    return (unsigned short)(r >> 16);
}

// ---------------------------------------------------------------------------
// Constants
// ---------------------------------------------------------------------------
#define NB   64      // batches
#define PN   2048    // continuous positions
#define LL   512     // byte sequence length
#define EE   256     // embed dim
#define MT   64      // P-tile per workgroup
#define KT   64      // K-tile
#define AST  72      // padded LDS row stride (elements) for A
#define BST  72      // padded LDS row stride (elements) for B^T

#define LOG10000_OVER_128 0.0719558158847f   // ln(10000)/128

// TDM availability (device pass only; host pass parses the fallback path)
#if defined(__HIP_DEVICE_COMPILE__) && __has_builtin(__builtin_amdgcn_tensor_load_to_lds)
#define HAVE_TDM 1
#else
#define HAVE_TDM 0
#endif

// ---------------------------------------------------------------------------
// Kernel 1: B^T[b][e][l] = bf16( emb_table[seq[b,l]][e] + PE(l,e) )
// grid = NB*EE blocks, 256 threads; each block owns one (b,e) row of length L.
// ---------------------------------------------------------------------------
__global__ __launch_bounds__(256) void embed_prep(
    const int*   __restrict__ seq,     // (NB, LL)
    const float* __restrict__ emb,     // (256, EE)
    unsigned short* __restrict__ BT)   // (NB, EE, LL) bf16
{
    const int b = blockIdx.x >> 8;
    const int e = blockIdx.x & 255;
    const int i = e >> 1;
    const float freq  = __expf((float)i * (-LOG10000_OVER_128));
    const bool  iscos = (e & 1);
    unsigned short* dst = BT + ((size_t)b * EE + e) * LL;
    for (int l = threadIdx.x; l < LL; l += 256) {
        int   byte = seq[b * LL + l];
        float ang  = (float)l * freq;
        float pe   = iscos ? __cosf(ang) : __sinf(ang);
        dst[l] = f2bf(emb[byte * EE + e] + pe);
    }
}

// ---------------------------------------------------------------------------
// Kernel 2: per (b,l) inverse width and amplitude
// ---------------------------------------------------------------------------
__global__ __launch_bounds__(256) void loc_prep(
    const int*   __restrict__ seq,     // (NB, LL)
    const float* __restrict__ loc,     // (256, 3)
    float* __restrict__ winv,          // (NB*LL)
    float* __restrict__ amp)           // (NB*LL)
{
    int idx = blockIdx.x * 256 + threadIdx.x;
    if (idx >= NB * LL) return;
    int byte = seq[idx];
    float w  = fabsf(loc[byte * 3 + 1]) * 0.02f + 1e-5f;
    winv[idx] = 1.0f / w;
    amp[idx]  = 1.0f / (1.0f + __expf(-loc[byte * 3 + 2]));
}

// ---------------------------------------------------------------------------
// Kernel 3: batched GEMM via v_wmma_f32_16x16x32_bf16
// grid = NB*32 blocks (32 P-tiles of 64), 256 threads = 8 waves.
// wave (wm = wave&3) -> M rows [wm*16, wm*16+16); (wn = wave>>2) -> N half.
// B^T K-tile is staged into LDS by the Tensor Data Mover (one tensor_load_to_lds
// per K-tile, pad_enable producing the stride-72 layout), overlapped with the
// Gaussian A-tile computation on the VALU.
// ---------------------------------------------------------------------------
__global__ __launch_bounds__(256) void field_gemm(
    const float*          __restrict__ xpos,  // (NB, PN)
    const unsigned short* __restrict__ BT,    // (NB, EE, LL) bf16
    const float*          __restrict__ winv,  // (NB, LL)
    const float*          __restrict__ amp,   // (NB, LL)
    float*                __restrict__ out)   // (NB, PN, EE)
{
    __shared__ __align__(16) unsigned short Alds[MT * AST];  //  9.0 KB
    __shared__ __align__(16) unsigned short Blds[EE * BST];  // 36.0 KB
    __shared__ float xl[MT];
    __shared__ float wvl[KT];
    __shared__ float aml[KT];

    const int b     = blockIdx.x >> 5;
    const int pt    = blockIdx.x & 31;
    const int pbase = pt * MT;
    const int tid   = (int)threadIdx.x;
    const int lane  = tid & 31;
    const int wave  = tid >> 5;
    const int mbase = (wave & 3) * 16;
    const int nbase = (wave >> 2) * 128;

    if (tid < MT) xl[tid] = xpos[b * PN + pbase + tid];

    v8f acc[8];
#pragma unroll
    for (int i = 0; i < 8; ++i) acc[i] = (v8f){0.f,0.f,0.f,0.f,0.f,0.f,0.f,0.f};

    const unsigned short* BTb = BT + (size_t)b * EE * LL;

    for (int kt = 0; kt < LL / KT; ++kt) {
        const int kbase = kt * KT;
        __syncthreads();   // previous tile fully consumed (also covers xl init)

#if HAVE_TDM
        // ---- TDM: async-stage B^T tile (256 rows x 64 bf16) with LDS padding
        if (wave == 0) {
            unsigned long long gaddr = (unsigned long long)(uintptr_t)(BTb + kbase);
            unsigned int ldsa = (unsigned int)(uintptr_t)(&Blds[0]);
            v4u g0;
            g0[0] = 1u;                                   // count=1 valid descriptor
            g0[1] = ldsa;                                 // lds_addr (bytes)
            g0[2] = (unsigned int)(gaddr & 0xFFFFFFFFu);  // global_addr[31:0]
            g0[3] = (unsigned int)((gaddr >> 32) & 0x01FFFFFFu) | (2u << 30); // addr[56:32] | type=2
            v8i g1;
            g1[0] = (1 << 16)        // data_size = 2 bytes
                  | (1 << 20)        // pad_enable
                  | (4 << 22)        // pad_interval: every 32 DWORDs (=64 bf16 = one row)
                  | (3 << 25);       // pad_amount: 4 DWORDs (=8 bf16) -> row stride 72
            g1[1] = (LL & 0xFFFF) << 16;       // tensor_dim0[15:0] = 512
            g1[2] = (EE << 16);                // tensor_dim0[31:16]=0 | tensor_dim1[15:0]=256
            g1[3] = (KT << 16);                // tensor_dim1[31:16]=0 | tile_dim0=64
            g1[4] = EE;                        // tile_dim1=256, tile_dim2=0
            g1[5] = LL;                        // tensor_dim0_stride[31:0] = 512 elements
            g1[6] = 0;                         // stride[47:32]=0 | tensor_dim1_stride lo=0
            g1[7] = 0;
            v4i g2 = {0, 0, 0, 0};
            v4i g3 = {0, 0, 0, 0};
#if defined(__clang_major__) && (__clang_major__ >= 23)
            v8i g4 = {0, 0, 0, 0, 0, 0, 0, 0};
            __builtin_amdgcn_tensor_load_to_lds(g0, g1, g2, g3, g4, 0);
#else
            __builtin_amdgcn_tensor_load_to_lds(g0, g1, g2, g3, 0);
#endif
        }
#else
        // ---- fallback: manual stage of B^T tile as 2048 16-byte chunks ----
#pragma unroll
        for (int i = 0; i < 8; ++i) {
            int c   = i * 256 + tid;
            int e   = c & 255;
            int seg = c >> 8;
            uint4 v = *(const uint4*)(BTb + (size_t)e * LL + kbase + seg * 8);
            *(uint4*)(&Blds[e * BST + seg * 8]) = v;
        }
#endif
        if (tid < KT) {
            wvl[tid] = winv[b * LL + kbase + tid];
            aml[tid] = amp [b * LL + kbase + tid];
        }
        __syncthreads();   // wvl/aml visible for A-tile compute

        // ---- compute A tile: 64 x 64 Gaussian basis, 16 elems/thread ----
        // (VALU work here overlaps the in-flight TDM transfer)
        {
            const int m  = tid >> 2;
            const int kq = (tid & 3) * 16;
            const float x = xl[m];
            unsigned int packed[8];
#pragma unroll
            for (int j = 0; j < 16; j += 2) {
                int   k0 = kq + j;
                float p0 = (float)(kbase + k0)     * (1.0f / 511.0f);
                float p1 = (float)(kbase + k0 + 1) * (1.0f / 511.0f);
                float d0 = (x - p0) * wvl[k0];
                float d1 = (x - p1) * wvl[k0 + 1];
                float v0 = __expf(-0.5f * d0 * d0) * aml[k0];
                float v1 = __expf(-0.5f * d1 * d1) * aml[k0 + 1];
                packed[j >> 1] = (unsigned)f2bf(v0) | ((unsigned)f2bf(v1) << 16);
            }
            uint4* dst = (uint4*)&Alds[m * AST + kq];
            dst[0] = make_uint4(packed[0], packed[1], packed[2], packed[3]);
            dst[1] = make_uint4(packed[4], packed[5], packed[6], packed[7]);
        }
#if HAVE_TDM
        if (wave == 0) __builtin_amdgcn_s_wait_tensorcnt(0);  // TDM writes landed
#endif
        __syncthreads();   // A + B tiles ready

        // ---- WMMA: hoist A fragments, double-buffer B fragments ----
        FragBF16 afr[2];
#pragma unroll
        for (int h = 0; h < 2; ++h) {
            // A frag: lane -> M = mbase + lane%16; K pairs at
            // {0,2,4,6,16,18,20,22} + 8*(lane/16) + 32*h  (ISA 16-bit A 16x32)
            const unsigned short* arow =
                &Alds[(mbase + (lane & 15)) * AST + h * 32 + (lane >> 4) * 8];
            uint4 lo = *(const uint4*)(arow);        // bytes [0,16)  -> u[0..3]
            uint4 hi = *(const uint4*)(arow + 16);   // bytes [32,48) -> u[4..7]
            afr[h].u[0] = lo.x; afr[h].u[1] = lo.y; afr[h].u[2] = lo.z; afr[h].u[3] = lo.w;
            afr[h].u[4] = hi.x; afr[h].u[5] = hi.y; afr[h].u[6] = hi.z; afr[h].u[7] = hi.w;
        }

        auto loadB = [&](int s, FragBF16& f) {
            // B frag: lane -> N = lane%16, K = 32*(s/8) + 8*(lane/16) + {0..7, 16..23}
            // (same K order as the A fragment).
            const int ks = (s >> 3) * 32;
            const int nt = s & 7;
            const unsigned short* brow =
                &Blds[(nbase + nt * 16 + (lane & 15)) * BST + ks + (lane >> 4) * 8];
            uint4 lo = *(const uint4*)(brow);
            uint4 hi = *(const uint4*)(brow + 16);
            f.u[0] = lo.x; f.u[1] = lo.y; f.u[2] = lo.z; f.u[3] = lo.w;
            f.u[4] = hi.x; f.u[5] = hi.y; f.u[6] = hi.z; f.u[7] = hi.w;
        };

        FragBF16 bb[2];
        loadB(0, bb[0]);
#pragma unroll
        for (int s = 0; s < 16; ++s) {
            if (s < 15) loadB(s + 1, bb[(s + 1) & 1]);
            acc[s & 7] = __builtin_amdgcn_wmma_f32_16x16x32_bf16(
                /*neg_a=*/false, afr[s >> 3].v, /*neg_b=*/false, bb[s & 1].v,
                /*c_mod=*/(short)0, acc[s & 7],
                /*reuse_a=*/false, /*reuse_b=*/false);
        }
    }

    // ---- epilogue: C/D layout: VGPR v -> M = v + 8*(lane/16), N = lane%16 ----
    // Output is 128 MB write-once: use non-temporal stores.
    float* outb = out + ((size_t)b * PN + pbase) * EE;
    const int mtop = mbase + (lane >> 4) * 8;
#pragma unroll
    for (int nt = 0; nt < 8; ++nt) {
        const int n = nbase + nt * 16 + (lane & 15);
#pragma unroll
        for (int v = 0; v < 8; ++v) {
            __builtin_nontemporal_store(acc[nt][v], &outb[(size_t)(mtop + v) * EE + n]);
        }
    }
}

// ---------------------------------------------------------------------------
// Fallback (only if workspace is too small): naive fused per-element kernel.
// ---------------------------------------------------------------------------
__global__ __launch_bounds__(256) void fused_naive(
    const int* __restrict__ seq, const float* __restrict__ xpos,
    const float* __restrict__ emb, const float* __restrict__ loc,
    float* __restrict__ out)
{
    size_t idx = (size_t)blockIdx.x * 256 + threadIdx.x;
    int e = (int)(idx & 255);
    int p = (int)((idx >> 8) & 2047);
    int b = (int)(idx >> 19);
    float x    = xpos[b * PN + p];
    float freq = __expf((float)(e >> 1) * (-LOG10000_OVER_128));
    float s = 0.f;
    for (int l = 0; l < LL; ++l) {
        int byte = seq[b * LL + l];
        float w = fabsf(loc[byte * 3 + 1]) * 0.02f + 1e-5f;
        float a = 1.f / (1.f + __expf(-loc[byte * 3 + 2]));
        float d = (x - (float)l * (1.0f / 511.0f)) / w;
        float basis = __expf(-0.5f * d * d) * a;
        float ang = (float)l * freq;
        float pe = (e & 1) ? __cosf(ang) : __sinf(ang);
        s += basis * (emb[byte * EE + e] + pe);
    }
    out[idx] = s;
}

// ---------------------------------------------------------------------------
// Host launch
// ---------------------------------------------------------------------------
extern "C" void kernel_launch(void* const* d_in, const int* in_sizes, int n_in,
                              void* d_out, int out_size, void* d_ws, size_t ws_size,
                              hipStream_t stream) {
    const int*   seq  = (const int*)  d_in[0];   // byte_sequence (64,512)
    const float* xpos = (const float*)d_in[1];   // continuous_positions (64,2048)
    const float* emb  = (const float*)d_in[2];   // emb_table (256,256)
    const float* loc  = (const float*)d_in[3];   // locality_shapes (256,3)
    float*       out  = (float*)d_out;           // (64,2048,2,128) flat

    const size_t BT_BYTES   = (size_t)NB * EE * LL * sizeof(unsigned short); // 16 MB
    const size_t WA_BYTES   = (size_t)NB * LL * sizeof(float);               // 128 KB
    const size_t WS_NEEDED  = BT_BYTES + 2 * WA_BYTES;

    if (ws_size >= WS_NEEDED) {
        unsigned short* BT   = (unsigned short*)d_ws;
        float*          winv = (float*)((char*)d_ws + BT_BYTES);
        float*          amp  = winv + (size_t)NB * LL;

        embed_prep<<<NB * EE, 256, 0, stream>>>(seq, emb, BT);
        loc_prep  <<<(NB * LL + 255) / 256, 256, 0, stream>>>(seq, loc, winv, amp);
        field_gemm<<<NB * 32, 256, 0, stream>>>(xpos, BT, winv, amp, out);
    } else {
        fused_naive<<<(NB * PN * EE) / 256, 256, 0, stream>>>(seq, xpos, emb, loc, out);
    }
}
